// GAT_27496380629162
// MI455X (gfx1250) — matrix operand order repaired and checked by reading, not compiled, not run on hardware
//
#include <hip/hip_runtime.h>
#include <hip/hip_bf16.h>

// ---------------------------------------------------------------------------
// GAT (2x GATConv + projection + softmax) for MI455X / gfx1250.
// GEMMs via v_wmma_f32_16x16x32_bf16 (wave32), edge softmax via L2 atomics.
// ---------------------------------------------------------------------------

typedef __attribute__((ext_vector_type(16))) __bf16 v16bf;
typedef __attribute__((ext_vector_type(8)))  float  v8f;

#define GAT_NN   50000
#define GAT_NE   800000
#define GAT_FIN  128   // both layers: 128 in, 128 out (H*D = 4*32)
#define GAT_NH   4
#define GAT_HD   32

static __device__ __forceinline__ __bf16 f2bf(float f) {
    unsigned u = __builtin_bit_cast(unsigned, f);
    unsigned r = u + 0x7FFFu + ((u >> 16) & 1u);   // round-to-nearest-even
    unsigned short h = (unsigned short)(r >> 16);
    return __builtin_bit_cast(__bf16, h);
}

// monotone encoding: float order -> unsigned order (for atomicMax on floats)
static __device__ __forceinline__ unsigned fenc(float f) {
    unsigned u = __builtin_bit_cast(unsigned, f);
    return (u & 0x80000000u) ? ~u : (u | 0x80000000u);
}
static __device__ __forceinline__ float fdec(unsigned u) {
    unsigned b = (u & 0x80000000u) ? (u & 0x7FFFFFFFu) : ~u;
    return __builtin_bit_cast(float, b);
}

// --------------------------- zero scratch ----------------------------------
__global__ void gat_zero_k(float* __restrict__ p, long long n) {
    long long i = (long long)blockIdx.x * blockDim.x + threadIdx.x;
    long long s = (long long)gridDim.x * blockDim.x;
    for (; i < n; i += s) p[i] = 0.0f;
}

// --------------------------- WMMA GEMM: Y[N,128] = X[N,128] @ W[128,128] ----
// Block: 256 threads = 8 waves; each wave owns a 16-row strip (block = 128 rows).
// W is staged in LDS pre-swizzled into this kernel's contiguous-k B-operand layout (32x16 bf16
// tiles): entry((kt,nt),lane,i) = W[kt*32 + 16*(lane>=16) + i][nt*16 + lane%16].
// Staging reads W coalesced (float4, row-major) and scatters 2B stores to LDS.
__global__ __launch_bounds__(256) void gat_gemm_wmma_k(
    const float* __restrict__ X, const float* __restrict__ W,
    float* __restrict__ Y, int nrows)
{
    __shared__ __align__(32) __bf16 sW[4 * 8 * 32 * 16];   // 32 KB

    int tid = threadIdx.x;
    const float4* W4 = (const float4*)W;
#pragma unroll 4
    for (int j = 0; j < 16; ++j) {
        int S4 = tid + j * 256;          // float4 index 0..4095 (coalesced)
        float4 w = W4[S4];
        int S    = S4 * 4;               // flat element index
        int k    = S >> 7;               // W row (K index)
        int colb = S & 127;              // starting W column
        int kt   = k >> 5;
        int kk   = k & 31;
        int half = (kk >= 16) ? 1 : 0;
        int i    = kk & 15;
        float wv[4] = {w.x, w.y, w.z, w.w};
#pragma unroll
        for (int c = 0; c < 4; ++c) {
            int col  = colb + c;
            int nt   = col >> 4;
            int lane = (col & 15) + half * 16;
            sW[(((kt * 8 + nt) * 32) + lane) * 16 + i] = f2bf(wv[c]);
        }
    }
    __syncthreads();

    int wid  = tid >> 5;
    int lane = tid & 31;
    int m0   = blockIdx.x * 128 + wid * 16;
    int hs   = (lane >= 16) ? 1 : 0;

    // A-operand: lane (l%16) holds row M=l%16; lo half-wave K 0..15,
    // hi half-wave K 16..31 (per 32-wide K tile).
    int rowA  = m0 + (lane & 15);
    int rowAc = rowA < nrows ? rowA : (nrows - 1);
    const float* xr = X + (long long)rowAc * GAT_FIN;

    v16bf a[4];
#pragma unroll
    for (int kt = 0; kt < 4; ++kt) {
        int base = kt * 32 + hs * 16;
#pragma unroll
        for (int i = 0; i < 8; ++i) a[kt][i]     = f2bf(xr[base + i]);
#pragma unroll
        for (int i = 0; i < 8; ++i) a[kt][8 + i] = f2bf(xr[base + 8 + i]);
    }

    const bool fullStrip = (m0 + 16) <= nrows;   // wave-uniform fast path

#pragma unroll
    for (int nt = 0; nt < 8; ++nt) {
        v8f c = {};
#pragma unroll
        for (int kt = 0; kt < 4; ++kt) {
            const v16bf* bp =
                (const v16bf*)&sW[(((kt * 8 + nt) * 32) + lane) * 16];
            c = __builtin_amdgcn_wmma_f32_16x16x32_bf16(
                    false, a[kt], false, *bp, (short)0, c, false, false);
        }
        // C/D layout: VGPR r, lanes 0-15 -> M=r, lanes 16-31 -> M=r+8
        int coln        = nt * 16 + (lane & 15);
        long long rbase = m0 + hs * 8;
        float* yp       = Y + rbase * GAT_FIN + coln;
        if (fullStrip) {
#pragma unroll
            for (int r = 0; r < 8; ++r) yp[r * GAT_FIN] = c[r];
        } else {
#pragma unroll
            for (int r = 0; r < 8; ++r)
                if (rbase + r < nrows) yp[r * GAT_FIN] = c[r];
        }
    }
}

// --------------------------- el/er = per-head dot(hf, al/ar) ----------------
__global__ void gat_eler_k(const float* __restrict__ hf,
                           const float* __restrict__ al,
                           const float* __restrict__ ar,
                           float* __restrict__ el, float* __restrict__ er,
                           int n)
{
    int id = blockIdx.x * blockDim.x + threadIdx.x;   // node*4 + h
    if (id >= n * GAT_NH) return;
    int h = id & 3, node = id >> 2;
    const float* row = hf + (long long)node * GAT_FIN + h * GAT_HD;
    const float* av  = al + h * GAT_HD;
    const float* rv  = ar + h * GAT_HD;
    float sl = 0.f, sr = 0.f;
#pragma unroll
    for (int d = 0; d < GAT_HD; ++d) { float v = row[d]; sl += v * av[d]; sr += v * rv[d]; }
    el[id] = sl; er[id] = sr;
}

// --------------------------- edge pass 1: segment max -----------------------
__global__ void gat_edge_max_k(const int* __restrict__ src,
                               const int* __restrict__ dst,
                               const float* __restrict__ el,
                               const float* __restrict__ er,
                               unsigned* __restrict__ emax, int ne)
{
    int id = blockIdx.x * blockDim.x + threadIdx.x;   // edge*4 + h
    if (id >= ne * GAT_NH) return;
    int h = id & 3, e = id >> 2;
    int s = src[e], d = dst[e];
    float v = el[s * 4 + h] + er[d * 4 + h];
    v = (v > 0.f) ? v : 0.2f * v;                     // leaky_relu(0.2)
    atomicMax(&emax[d * 4 + h], fenc(v));
}

// --------------------------- edge pass 2: numerator + denominator -----------
// One wave32 per edge; lane = dim within head. Atomics resolve in L2.
__global__ __launch_bounds__(256) void gat_edge_acc_k(
    const int* __restrict__ src, const int* __restrict__ dst,
    const float* __restrict__ el, const float* __restrict__ er,
    const unsigned* __restrict__ emax, const float* __restrict__ hf,
    float* __restrict__ num, float* __restrict__ denom, int ne)
{
    int gw   = (int)((blockIdx.x * blockDim.x + threadIdx.x) >> 5);
    int lane = threadIdx.x & 31;
    if (gw >= ne) return;
    int s = src[gw], d = dst[gw];
#pragma unroll
    for (int h = 0; h < GAT_NH; ++h) {
        float e = el[s * 4 + h] + er[d * 4 + h];
        e = (e > 0.f) ? e : 0.2f * e;
        float m  = fdec(emax[d * 4 + h]);
        float ex = __expf(e - m);
        float v  = ex * hf[(long long)s * GAT_FIN + h * GAT_HD + lane];
        atomicAdd(&num[(long long)d * GAT_FIN + h * GAT_HD + lane], v);
        if (lane == 0) atomicAdd(&denom[d * 4 + h], ex);
    }
}

// --------------------------- layer-1 finalize: h = relu(num/denom + b) ------
__global__ void gat_fin_relu_k(float* __restrict__ numh,
                               const float* __restrict__ denom,
                               const float* __restrict__ bias, int n)
{
    int id = blockIdx.x * blockDim.x + threadIdx.x;
    if (id >= n * GAT_FIN) return;
    int j = id & 127, node = id >> 7, h = j >> 5;
    float v = numh[id] / fmaxf(denom[node * 4 + h], 1e-30f) + bias[j];
    numh[id] = fmaxf(v, 0.0f);
}

// --------------------------- layer-2 finalize: head-mean -> proj -> softmax -
__global__ void gat_fin_out_k(const float* __restrict__ num,
                              const float* __restrict__ denom,
                              const float* __restrict__ b2,
                              const float* __restrict__ Wout,
                              const float* __restrict__ bout,
                              float* __restrict__ out, int n)
{
    int node = blockIdx.x * blockDim.x + threadIdx.x;
    if (node >= n) return;
    float dn[GAT_NH];
#pragma unroll
    for (int h = 0; h < GAT_NH; ++h) dn[h] = fmaxf(denom[node * 4 + h], 1e-30f);
    float l0 = bout[0], l1 = bout[1];
    const float* nr = num + (long long)node * GAT_FIN;
    for (int d = 0; d < GAT_HD; ++d) {
        float s = 0.f;
#pragma unroll
        for (int h = 0; h < GAT_NH; ++h)
            s += nr[h * GAT_HD + d] / dn[h] + b2[h * GAT_HD + d];
        float r = fmaxf(s * 0.25f, 0.0f);            // mean over heads, relu
        l0 += r * Wout[d * 2 + 0];
        l1 += r * Wout[d * 2 + 1];
    }
    float m  = fmaxf(l0, l1);
    float e0 = __expf(l0 - m), e1 = __expf(l1 - m);
    float inv = 1.0f / (e0 + e1);
    out[node * 2 + 0] = e0 * inv;
    out[node * 2 + 1] = e1 * inv;
}

// ---------------------------------------------------------------------------
extern "C" void kernel_launch(void* const* d_in, const int* in_sizes, int n_in,
                              void* d_out, int out_size, void* d_ws, size_t ws_size,
                              hipStream_t stream) {
    const float* in_feat = (const float*)d_in[0];
    const float* W1   = (const float*)d_in[1];
    const float* al1  = (const float*)d_in[2];
    const float* ar1  = (const float*)d_in[3];
    const float* b1   = (const float*)d_in[4];
    const float* W2   = (const float*)d_in[5];
    const float* al2  = (const float*)d_in[6];
    const float* ar2  = (const float*)d_in[7];
    const float* b2   = (const float*)d_in[8];
    const float* Wout = (const float*)d_in[9];
    const float* bout = (const float*)d_in[10];
    const int*   src  = (const int*)d_in[11];
    const int*   dst  = (const int*)d_in[12];
    float* out = (float*)d_out;

    const int NN = GAT_NN;
    const int NE = (n_in > 11) ? in_sizes[11] : GAT_NE;

    // workspace carve (contiguous): bufA | bufB | el | er | emax | denom
    float*    bufA  = (float*)d_ws;                 // hf (both layers)  N*128
    float*    bufB  = bufA + (long long)NN * 128;   // num / h           N*128
    float*    el    = bufB + (long long)NN * 128;   // N*4
    float*    er    = el + NN * 4;                  // N*4
    unsigned* emax  = (unsigned*)(er + NN * 4);     // N*4
    float*    denom = (float*)(emax + NN * 4);      // N*4
    const long long zlen = (long long)NN * (128 + 4 + 4 + 4 + 4); // bufB..denom

    dim3 b256(256);
    int gGemm = (NN + 127) / 128;
    int gElEr = (NN * 4 + 255) / 256;
    int gEmax = (NE * 4 + 255) / 256;
    int gEacc = (NE * 32 + 255) / 256;
    int gFin1 = (NN * 128 + 255) / 256;
    int gFin2 = (NN + 255) / 256;

    // ---------------- layer 1 ----------------
    gat_zero_k<<<2048, b256, 0, stream>>>(bufB, zlen);
    gat_gemm_wmma_k<<<gGemm, b256, 0, stream>>>(in_feat, W1, bufA, NN);
    gat_eler_k<<<gElEr, b256, 0, stream>>>(bufA, al1, ar1, el, er, NN);
    gat_edge_max_k<<<gEmax, b256, 0, stream>>>(src, dst, el, er, emax, NE);
    gat_edge_acc_k<<<gEacc, b256, 0, stream>>>(src, dst, el, er, emax, bufA,
                                               bufB, denom, NE);
    gat_fin_relu_k<<<gFin1, b256, 0, stream>>>(bufB, denom, b1, NN);  // bufB = h

    // ---------------- layer 2 ----------------
    gat_gemm_wmma_k<<<gGemm, b256, 0, stream>>>(bufB, W2, bufA, NN);  // hf2=bufA
    gat_zero_k<<<2048, b256, 0, stream>>>(bufB, zlen);                // num2/emax/denom
    gat_eler_k<<<gElEr, b256, 0, stream>>>(bufA, al2, ar2, el, er, NN);
    gat_edge_max_k<<<gEmax, b256, 0, stream>>>(src, dst, el, er, emax, NE);
    gat_edge_acc_k<<<gEacc, b256, 0, stream>>>(src, dst, el, er, emax, bufA,
                                               bufB, denom, NE);
    gat_fin_out_k<<<gFin2, b256, 0, stream>>>(bufB, denom, b2, Wout, bout,
                                              out, NN);
}
